// DetCenterSparse_7370163880520
// MI455X (gfx1250) — compile-verified, matrix operand order checked by reading, not run on hardware
//
#include <hip/hip_runtime.h>
#include <hip/hip_bf16.h>

#define NEG_INF_F (-1000000000.0f)

typedef __attribute__((ext_vector_type(16))) __bf16 v16bf;
typedef __attribute__((ext_vector_type(8)))  float  v8f;

// ---------------------------------------------------------------------------
// helpers
// ---------------------------------------------------------------------------
static __device__ __forceinline__ unsigned int f2bf_bits(float f) {
    unsigned int u = __float_as_uint(f);
    unsigned int r = u + 0x7FFFu + ((u >> 16) & 1u);   // round-to-nearest-even
    return r >> 16;
}

static __device__ __forceinline__ float wcat(const float* __restrict__ Wc,
                                             const float* __restrict__ Wr,
                                             int k, int n) {
    // fused weight matrix 256x16: cols 0-1 = W_cls, cols 2-10 = W_reg, 11-15 = 0
    if (n < 2)  return Wc[k * 2 + n];
    if (n < 11) return Wr[k * 9 + (n - 2)];
    return 0.0f;
}

static __device__ __forceinline__ unsigned int score_key(float s, bool inb) {
    if (!inb) return 0u;                               // below any real score
    unsigned int u = __float_as_uint(s);
    return (u & 0x80000000u) ? ~u : (u | 0x80000000u); // orderable float
}

// ---------------------------------------------------------------------------
// Kernel 1: pack fused weights into the bf16 WMMA B-matrix lane layout + bias
//   B layout (16-bit B 32x16, wave32): lane<16 -> col=lane, K offsets
//   {0..7,16..23}; lane>=16 -> col=lane-16, K offsets {8..15,24..31}.
//   Stored as Bpack[kstep][lane][8 dwords] so each lane reads 32B contiguous.
// ---------------------------------------------------------------------------
__global__ __launch_bounds__(256) void prep_kernel(
    const float* __restrict__ Wc, const float* __restrict__ bc,
    const float* __restrict__ Wr, const float* __restrict__ br,
    unsigned int* __restrict__ Bpack, float* __restrict__ bias16) {
    const int t = threadIdx.x;
    const int s = t >> 5;           // k-step 0..7
    const int lane = t & 31;
    const int n = lane & 15;
    const int khalf = (lane < 16) ? 0 : 8;
#pragma unroll
    for (int v = 0; v < 8; ++v) {
        int kb = s * 32 + ((v & 4) ? 16 : 0) + khalf + (v & 3) * 2;
        unsigned int lo = f2bf_bits(wcat(Wc, Wr, kb, n));
        unsigned int hi = f2bf_bits(wcat(Wc, Wr, kb + 1, n));
        Bpack[(s * 32 + lane) * 8 + v] = (hi << 16) | lo;
    }
    if (t < 16) bias16[t] = (t < 2) ? bc[t] : ((t < 11) ? br[t - 2] : 0.0f);
}

// ---------------------------------------------------------------------------
// Kernel 2: fused GEMM (WMMA bf16) + box decode.
//   Block = 256 threads = 8 waves; each wave owns one 16-row tile.
//   No __syncthreads (tail waves may exit early) — per-wave LDS transpose
//   guarded by s_wait_dscnt.
// ---------------------------------------------------------------------------
__global__ __launch_bounds__(256) void gemm_decode_kernel(
    const float* __restrict__ feat, const int* __restrict__ coor,
    const unsigned int* __restrict__ Bpack, const float* __restrict__ bias16,
    float* __restrict__ box8, float* __restrict__ scores, int nrows) {
    __shared__ float tile[8][16][16];

    const int lane = threadIdx.x & 31;
    const int w    = threadIdx.x >> 5;
    const int m0   = (blockIdx.x * 8 + w) * 16;
    if (m0 >= nrows) return;                    // uniform within wave

    // ---- load pre-packed B (all 8 k-steps, 32B contiguous per lane) ----
    v16bf bmat[8];
#pragma unroll
    for (int s = 0; s < 8; ++s)
        bmat[s] = *(const v16bf*)(Bpack + (s * 32 + lane) * 8);

    const float bias = bias16[lane & 15];
    v8f c;
#pragma unroll
    for (int r = 0; r < 8; ++r) c[r] = bias;

    // ---- A tile: lane<16 holds row m0+lane K{0..7,16..23}+32s,
    //              lane>=16 same row K{8..15,24..31}+32s ----
    const int mrow  = m0 + (lane & 15);
    const int khalf = (lane < 16) ? 0 : 8;
    const float4* rp = (const float4*)(feat + (size_t)mrow * 256 + khalf);

#pragma unroll
    for (int s = 0; s < 8; ++s) {
        float4 f0 = rp[s * 8 + 0];
        float4 f1 = rp[s * 8 + 1];
        float4 f2 = rp[s * 8 + 4];
        float4 f3 = rp[s * 8 + 5];
        v16bf a;
        a[0]  = (__bf16)f0.x; a[1]  = (__bf16)f0.y;
        a[2]  = (__bf16)f0.z; a[3]  = (__bf16)f0.w;
        a[4]  = (__bf16)f1.x; a[5]  = (__bf16)f1.y;
        a[6]  = (__bf16)f1.z; a[7]  = (__bf16)f1.w;
        a[8]  = (__bf16)f2.x; a[9]  = (__bf16)f2.y;
        a[10] = (__bf16)f2.z; a[11] = (__bf16)f2.w;
        a[12] = (__bf16)f3.x; a[13] = (__bf16)f3.y;
        a[14] = (__bf16)f3.z; a[15] = (__bf16)f3.w;
        c = __builtin_amdgcn_wmma_f32_16x16x32_bf16(
                false, a, false, bmat[s], (short)0, c, false, false);
    }

    // ---- transpose C through LDS (per-wave; DS ops in-order per wave) ----
    const int rbase = (lane < 16) ? 0 : 8;
    const int ncol  = lane & 15;
#pragma unroll
    for (int r = 0; r < 8; ++r) tile[w][rbase + r][ncol] = c[r];
    asm volatile("s_wait_dscnt 0" ::: "memory");

    if (lane < 16) {
        const int m = m0 + lane;
        float o[16];
#pragma unroll
        for (int j = 0; j < 16; ++j) o[j] = tile[w][lane][j];

        // softmax over 2 logits -> probs[:,1]
        float conf  = 1.0f / (1.0f + __expf(o[0] - o[1]));
        float scr   = 1.0f / (1.0f + __expf(-o[10]));
        float score = conf * scr;

        float cx = (float)coor[2 * m + 0] * 0.8f + o[2];
        float cy = (float)coor[2 * m + 1] * 0.8f + o[3];
        float z  = o[4];
        float el = __expf(fminf(fmaxf(o[5], -6.0f), 6.0f));
        float ew = __expf(fminf(fmaxf(o[6], -6.0f), 6.0f));
        float eh = __expf(fminf(fmaxf(o[7], -6.0f), 6.0f));
        float ang = atan2f(o[8], o[9]);

        float4 lo4 = {cx, cy, z, el};
        float4 hi4 = {ew, eh, ang, score};
        *(float4*)(box8 + (size_t)m * 8)     = lo4;
        *(float4*)(box8 + (size_t)m * 8 + 4) = hi4;
        scores[m] = score;
    }
}

// ---------------------------------------------------------------------------
// Kernel 3: per-batch exact top-1024 via 8-bit radix select + bitonic sort.
//   One 1024-thread block per batch. Sort key = (orderable_score, ~idx) so
//   output is descending by score, ties by lower index (matches lax.top_k).
// ---------------------------------------------------------------------------
__global__ __launch_bounds__(1024) void topk_kernel(
    const float* __restrict__ scores, const int* __restrict__ batch_idx,
    int N, int* __restrict__ topk_idx, float* __restrict__ topk_val) {
    const int b = blockIdx.x, tid = threadIdx.x;
    __shared__ unsigned int hist[256];
    __shared__ unsigned int s_sel, s_kth;
    __shared__ int s_cgt, s_ceq;
    __shared__ unsigned long long keys[1024];

    unsigned int prefix = 0, prefmask = 0, kth = 1024;
#pragma unroll 1
    for (int d = 3; d >= 0; --d) {
        if (tid < 256) hist[tid] = 0;
        __syncthreads();
        for (int i = tid; i < N; i += 1024) {
            unsigned int k = score_key(scores[i], batch_idx[i] == b);
            if ((k & prefmask) == prefix)
                atomicAdd(&hist[(k >> (d * 8)) & 255u], 1u);
        }
        __syncthreads();
        if (tid == 0) {
            unsigned int cum = 0, sel = 0, kk = kth;
            for (int bin = 255; bin >= 0; --bin) {
                unsigned int h = hist[bin];
                if (cum + h >= kk) { sel = (unsigned int)bin; kk -= cum; break; }
                cum += h;
            }
            s_sel = sel; s_kth = kk;
        }
        __syncthreads();
        prefix  |= s_sel << (d * 8);
        prefmask |= 255u << (d * 8);
        kth = s_kth;
        __syncthreads();
    }

    const unsigned int T = prefix;               // 1024th-largest key value
    if (tid == 0) { s_cgt = 0; s_ceq = 0; }
    __syncthreads();
    const int gtTotal = 1024 - (int)kth;
    for (int i = tid; i < N; i += 1024) {
        unsigned int k = score_key(scores[i], batch_idx[i] == b);
        unsigned long long key64 =
            ((unsigned long long)k << 32) | (unsigned int)(~(unsigned int)i);
        if (k > T) {
            int p = atomicAdd(&s_cgt, 1);
            keys[p] = key64;
        } else if (k == T) {
            int p = atomicAdd(&s_ceq, 1);
            if (p < (int)kth) keys[gtTotal + p] = key64;
        }
    }
    __syncthreads();

    // bitonic sort, descending
    for (unsigned int size = 2; size <= 1024; size <<= 1) {
        for (unsigned int stride = size >> 1; stride > 0; stride >>= 1) {
            unsigned int i = (unsigned int)tid, j = i ^ stride;
            if (j > i) {
                unsigned long long a = keys[i], cc = keys[j];
                bool descBlock = ((i & size) == 0);
                if (descBlock ? (a < cc) : (a > cc)) { keys[i] = cc; keys[j] = a; }
            }
            __syncthreads();
        }
    }

    unsigned long long kk = keys[tid];
    int idx = (int)(~(unsigned int)(kk & 0xFFFFFFFFull));
    topk_idx[b * 1024 + tid] = idx;
    topk_val[b * 1024 + tid] =
        (batch_idx[idx] == b) ? scores[idx] : NEG_INF_F;
}

// ---------------------------------------------------------------------------
// Kernel 4: greedy BEV NMS (sequential, barrier-stepped) + output emission.
// ---------------------------------------------------------------------------
__global__ __launch_bounds__(1024) void nms_out_kernel(
    const float* __restrict__ box8, const int* __restrict__ topk_idx,
    const float* __restrict__ topk_val, float* __restrict__ out,
    int* __restrict__ lb, unsigned char* __restrict__ keep_out) {
    const int b = blockIdx.x, t = threadIdx.x;
    __shared__ float sx1[1024], sx2[1024], sy1[1024], sy2[1024], sarea[1024];
    __shared__ unsigned char ssup[1024];

    const int   idx = topk_idx[b * 1024 + t];
    const float val = topk_val[b * 1024 + t];
    const float* bx = box8 + (size_t)idx * 8;
    float x = bx[0], y = bx[1], l = bx[3], wd = bx[4];
    sx1[t] = x - l * 0.5f;  sx2[t] = x + l * 0.5f;
    sy1[t] = y - wd * 0.5f; sy2[t] = y + wd * 0.5f;
    sarea[t] = l * wd;
    ssup[t] = 0;
    __syncthreads();

#pragma unroll 1
    for (int i = 0; i < 1024; ++i) {
        if (!ssup[i] && t > i) {
            float ix = fmaxf(fminf(sx2[i], sx2[t]) - fmaxf(sx1[i], sx1[t]), 0.0f);
            float iy = fmaxf(fminf(sy2[i], sy2[t]) - fmaxf(sy1[i], sy1[t]), 0.0f);
            float inter = ix * iy;
            float uni = sarea[i] + sarea[t] - inter;
            if (inter / fmaxf(uni, 1e-6f) > 0.1f) ssup[t] = 1;
        }
        __syncthreads();
    }

    const bool valid = val > (NEG_INF_F * 0.5f);
    const bool kp = (!ssup[t]) && valid;
    const float kf = kp ? 1.0f : 0.0f;

    float4 a0 = ((const float4*)bx)[0];   // x,y,z,l
    float4 a1 = ((const float4*)bx)[1];   // w,h,angle,score
    float* op = out + ((size_t)b * 1024 + t) * 8;
    float4 o0 = {a0.x * kf, a0.y * kf, a0.z * kf, a0.w * kf};
    float4 o1 = {a1.x * kf, a1.y * kf, a1.z * kf, val * kf};
    ((float4*)op)[0] = o0;
    ((float4*)op)[1] = o1;
    lb[b * 1024 + t] = 0;                 // cls_prob has 1 column -> argmax==0
    keep_out[b * 1024 + t] = kp ? 1 : 0;
}

// ---------------------------------------------------------------------------
// launcher
// ---------------------------------------------------------------------------
extern "C" void kernel_launch(void* const* d_in, const int* in_sizes, int n_in,
                              void* d_out, int out_size, void* d_ws, size_t ws_size,
                              hipStream_t stream) {
    const float* feat      = (const float*)d_in[0];
    const int*   batch_idx = (const int*)d_in[1];
    const int*   coor      = (const int*)d_in[2];
    const float* Wc        = (const float*)d_in[3];
    const float* bc        = (const float*)d_in[4];
    const float* Wr        = (const float*)d_in[5];
    const float* br        = (const float*)d_in[6];
    const int N = in_sizes[1];                 // 200000

    // workspace layout (32-bit units)
    unsigned int* Bpack = (unsigned int*)d_ws;                 // 8*32*8 = 2048
    float* bias16   = (float*)d_ws + 2048;                     // 16
    float* scores   = (float*)d_ws + 2064;                     // N
    float* box8     = (float*)d_ws + 2064 + (size_t)N;         // N*8 (16B aligned)
    int*   topk_idx = (int*)((float*)d_ws + 2064 + (size_t)N * 9);
    float* topk_val = (float*)(topk_idx + 4 * 1024);

    float*         out  = (float*)d_out;                       // 4*1024*8
    int*           lbp  = (int*)d_out + 32768;                 // 4*1024 int32
    unsigned char* keep = (unsigned char*)d_out + (size_t)36864 * 4;

    prep_kernel<<<1, 256, 0, stream>>>(Wc, bc, Wr, br, Bpack, bias16);

    int tiles  = (N + 15) / 16;
    int blocks = (tiles + 7) / 8;
    gemm_decode_kernel<<<blocks, 256, 0, stream>>>(feat, coor, Bpack, bias16,
                                                   box8, scores, N);

    topk_kernel<<<4, 1024, 0, stream>>>(scores, batch_idx, N, topk_idx, topk_val);
    nms_out_kernel<<<4, 1024, 0, stream>>>(box8, topk_idx, topk_val, out, lbp, keep);
}
